// SparseMNISTNet_59141699666075
// MI455X (gfx1250) — matrix operand order repaired and checked by reading, not compiled, not run on hardware
//
#include <hip/hip_runtime.h>
#include <hip/hip_bf16.h>

// ---- problem constants (match reference) ----
#define BATCH   16384
#define IN_DIM  784
#define KPAD    800          // IN_DIM padded to multiple of 32 for WMMA K-steps
#define NHID    2000
#define KWIN    200
#define ROWS    16           // batch rows per workgroup
#define THREADS 256
#define WAVES   8            // wave32
#define NTILES  (NHID / 16)  // 125 column tiles of 16
#define KSTEPS  (KPAD / 32)  // 25

typedef __attribute__((ext_vector_type(16))) __bf16 v16bf;
typedef __attribute__((ext_vector_type(8)))  float  v8f;

union ABFrag { v16bf v; unsigned u[8]; uint4 q[2]; };

// float -> bf16 with round-to-nearest-even
__device__ __forceinline__ unsigned short f2bf(float f) {
    unsigned u = __float_as_uint(f);
    unsigned r = u + 0x7FFFu + ((u >> 16) & 1u);
    return (unsigned short)(r >> 16);
}
__device__ __forceinline__ unsigned pack2bf(float lo, float hi) {
    return ((unsigned)f2bf(hi) << 16) | (unsigned)f2bf(lo);
}

// monotone transform: larger float -> larger unsigned key
__device__ __forceinline__ unsigned fkey(float f) {
    unsigned u = __float_as_uint(f);
    return (u & 0x80000000u) ? ~u : (u | 0x80000000u);
}

// pack 16 consecutive f32 (4x float4) into a bf16 B fragment half
__device__ __forceinline__ void packB16(ABFrag& b, const float4* p) {
    float4 f0 = p[0], f1 = p[1], f2 = p[2], f3 = p[3];
    b.u[0] = pack2bf(f0.x, f0.y); b.u[1] = pack2bf(f0.z, f0.w);
    b.u[2] = pack2bf(f1.x, f1.y); b.u[3] = pack2bf(f1.z, f1.w);
    b.u[4] = pack2bf(f2.x, f2.y); b.u[5] = pack2bf(f2.z, f2.w);
    b.u[6] = pack2bf(f3.x, f3.y); b.u[7] = pack2bf(f3.z, f3.w);
}

// ---- pre-pass: W1 f32 [2000 x 784] -> bf16 [2000 x 800], zero-padded ----
__global__ void conv_w1_bf16(const float* __restrict__ W1,
                             unsigned short* __restrict__ W1b) {
    int idx = blockIdx.x * THREADS + threadIdx.x;
    if (idx >= NHID * KPAD) return;
    int n = idx / KPAD;
    int k = idx - n * KPAD;
    W1b[idx] = (k < IN_DIM) ? f2bf(W1[(size_t)n * IN_DIM + k]) : (unsigned short)0;
}

template <bool PRECONV>
__global__ __launch_bounds__(THREADS, 1)
void sparse_mnist_fused(const float* __restrict__ x,
                        const float* __restrict__ W1,
                        const unsigned short* __restrict__ W1b,
                        const float* __restrict__ bias1,
                        const float* __restrict__ W2,
                        const float* __restrict__ bias2,
                        const float* __restrict__ duty,
                        float* __restrict__ out)
{
    __shared__ __align__(16) unsigned short xs[ROWS * KPAD];  // 25.6 KB bf16 A-tile
    __shared__ float    hs[ROWS * NHID];                      // 128 KB fused h tile
    __shared__ float    boosts[NHID];                         // 8 KB
    __shared__ unsigned hist[WAVES * 256];                    // 8 KB radix bins

    const int tid  = threadIdx.x;
    const int lane = tid & 31;
    const int wv   = tid >> 5;
    const int row0 = blockIdx.x * ROWS;

    // ---- stage x rows as bf16 (zero-padded K), precompute boost factors ----
    for (int idx = tid; idx < ROWS * KPAD; idx += THREADS) {
        int m = idx / KPAD;
        int k = idx - m * KPAD;
        float v = (k < IN_DIM) ? x[(size_t)(row0 + m) * IN_DIM + k] : 0.0f;
        xs[idx] = f2bf(v);
    }
    for (int j = tid; j < NHID; j += THREADS)
        boosts[j] = __expf(0.1f - duty[j]);   // targetDensity=0.1, boostStrength=1
    __syncthreads();

    // ---- GEMM1: h = x @ W1^T + b1 via v_wmma_f32_16x16x32_bf16 ----
    const int hl = lane >> 4;   // 16-lane half
    const int nn = lane & 15;   // M index for A, N index for B/C
    // A fragment base: u[0..3] and u[4..7] are each 16 contiguous bytes in LDS
    const unsigned short* xsp = &xs[nn * KPAD + hl * 16];

    for (int t = wv; t < NTILES; t += WAVES) {
        const int col0 = t * 16;
        v8f acc = {};

        if (PRECONV) {
            const unsigned short* __restrict__ w1bp =
                W1b + (size_t)(col0 + nn) * KPAD + hl * 16;
            {   // prefetch next tile's weight row
                int tn = t + WAVES;
                if (tn < NTILES)
                    __builtin_prefetch(W1b + (size_t)(tn * 16 + nn) * KPAD, 0, 0);
            }
#pragma unroll 5
            for (int ks = 0; ks < KSTEPS; ++ks) {
                const int k0 = ks * 32;
                ABFrag a, b;
                a.q[0] = *reinterpret_cast<const uint4*>(xsp + k0);        // ds_load_b128
                a.q[1] = *reinterpret_cast<const uint4*>(xsp + k0 + 8);    // ds_load_b128
                b.q[0] = *reinterpret_cast<const uint4*>(w1bp + k0);       // global_load_b128
                b.q[1] = *reinterpret_cast<const uint4*>(w1bp + k0 + 8);   // global_load_b128
                acc = __builtin_amdgcn_wmma_f32_16x16x32_bf16(
                    false, a.v, false, b.v, (short)0, acc, false, false);
            }
        } else {
            // fallback: read f32 W1 directly; 24 guard-free steps + uniform tail
            const float* __restrict__ w1p =
                W1 + (size_t)(col0 + nn) * IN_DIM + hl * 16;
#pragma unroll 4
            for (int ks = 0; ks < KSTEPS - 1; ++ks) {
                const int k0 = ks * 32;
                ABFrag a, b;
                a.q[0] = *reinterpret_cast<const uint4*>(xsp + k0);
                a.q[1] = *reinterpret_cast<const uint4*>(xsp + k0 + 8);
                packB16(b, reinterpret_cast<const float4*>(w1p + k0));
                acc = __builtin_amdgcn_wmma_f32_16x16x32_bf16(
                    false, a.v, false, b.v, (short)0, acc, false, false);
            }
            {   // tail K-step: k0=768; only hl==0 lanes have valid weights
                const int k0 = (KSTEPS - 1) * 32;
                ABFrag a, b;
                a.q[0] = *reinterpret_cast<const uint4*>(xsp + k0);
                a.q[1] = *reinterpret_cast<const uint4*>(xsp + k0 + 8);
                if (hl == 0) {
                    packB16(b, reinterpret_cast<const float4*>(w1p + k0));
                } else {
#pragma unroll
                    for (int i = 0; i < 8; ++i) b.u[i] = 0u;
                }
                acc = __builtin_amdgcn_wmma_f32_16x16x32_bf16(
                    false, a.v, false, b.v, (short)0, acc, false, false);
            }
        }

        const float bias = bias1[col0 + nn];
#pragma unroll
        for (int r = 0; r < 8; ++r) {
            int m = r + hl * 8;                 // C/D layout
            hs[m * NHID + col0 + nn] = acc[r] + bias;
        }
    }
    __syncthreads();

    // ---- per-row: exact top-K threshold (4-pass radix select) + mask,
    //      then GEMM2 (10 classes) + log_softmax. 2 rows per wave. ----
    unsigned* h8 = &hist[wv * 256];
    for (int rr = 0; rr < ROWS / WAVES; ++rr) {
        const int r = wv * (ROWS / WAVES) + rr;
        float* __restrict__ hrow = &hs[r * NHID];

        unsigned rem = KWIN;
        unsigned prefix = 0;
        for (int p = 0; p < 4; ++p) {
            const int shift = 24 - 8 * p;
            for (int bi = lane; bi < 256; bi += 32) h8[bi] = 0;
            __syncthreads();
            for (int j = lane; j < NHID; j += 32) {
                unsigned key = fkey(hrow[j] * boosts[j]);
                bool match = (p == 0) || ((key >> (shift + 8)) == prefix);
                if (match) atomicAdd(&h8[(key >> shift) & 255u], 1u);
            }
            __syncthreads();

            // ---- parallel bin select: lane L owns bins [8L .. 8L+7] ----
            unsigned c[8];
            unsigned Sl = 0;
#pragma unroll
            for (int i = 0; i < 8; ++i) { c[i] = h8[lane * 8 + i]; Sl += c[i]; }
            // inclusive suffix-scan across lanes: incl = sum_{l' >= lane} S_{l'}
            unsigned incl = Sl;
#pragma unroll
            for (int s = 1; s < 32; s <<= 1) {
                unsigned v = __shfl_down(incl, s, 32);
                if (lane + s < 32) incl += v;
            }
            const unsigned above = incl - Sl;       // count in bins above this lane's
            const bool has = (above < rem) && (incl >= rem);
            unsigned bsel = 0, cumsel = 0;
            if (has) {
                unsigned cum = above;
#pragma unroll
                for (int i = 7; i >= 0; --i) {
                    unsigned ci = c[i];
                    if (cum + ci >= rem) { bsel = (unsigned)(lane * 8 + i); cumsel = cum; break; }
                    cum += ci;
                }
            }
            unsigned long long bm = __ballot(has);  // exactly one lane set
            int src = __ffsll((unsigned long long)bm) - 1;
            bsel   = __shfl(bsel,   src, 32);
            cumsel = __shfl(cumsel, src, 32);
            prefix = (prefix << 8) | bsel;
            rem -= cumsel;
        }
        const unsigned T = prefix;  // exact key of the 200th-largest boosted value

        // keep original h where boosted key >= threshold, else zero
        for (int j = lane; j < NHID; j += 32) {
            float h = hrow[j];
            unsigned key = fkey(h * boosts[j]);
            hrow[j] = (key >= T) ? h : 0.0f;
        }

        // GEMM2: logits = h_sparse @ W2^T + b2 ; then log_softmax
        float logits[10];
        float mx = -3.0e38f;
#pragma unroll
        for (int c2 = 0; c2 < 10; ++c2) {
            const float* __restrict__ w2row = W2 + c2 * NHID;
            float acc = 0.0f;
            for (int j = lane; j < NHID; j += 32)
                acc += hrow[j] * w2row[j];
#pragma unroll
            for (int s = 16; s > 0; s >>= 1)
                acc += __shfl_xor(acc, s, 32);
            acc += bias2[c2];
            logits[c2] = acc;
            mx = fmaxf(mx, acc);
        }
        float se = 0.0f;
#pragma unroll
        for (int c2 = 0; c2 < 10; ++c2) se += __expf(logits[c2] - mx);
        const float lse = __logf(se);
        if (lane == 0) {
            float* o = out + (size_t)(row0 + r) * 10;
#pragma unroll
            for (int c2 = 0; c2 < 10; ++c2) o[c2] = logits[c2] - mx - lse;
        }
    }
}

extern "C" void kernel_launch(void* const* d_in, const int* in_sizes, int n_in,
                              void* d_out, int out_size, void* d_ws, size_t ws_size,
                              hipStream_t stream) {
    (void)in_sizes; (void)n_in; (void)out_size;
    const float* x     = (const float*)d_in[0];
    const float* W1    = (const float*)d_in[1];
    const float* b1    = (const float*)d_in[2];
    const float* W2    = (const float*)d_in[3];
    const float* b2    = (const float*)d_in[4];
    const float* duty  = (const float*)d_in[5];
    float* out = (float*)d_out;

    dim3 grid(BATCH / ROWS);   // 1024 workgroups x 16 rows
    dim3 block(THREADS);       // 8 wave32 waves

    const size_t need = (size_t)NHID * KPAD * sizeof(unsigned short);  // 3.2 MB
    if (ws_size >= need && d_ws != nullptr) {
        unsigned short* W1b = (unsigned short*)d_ws;
        int nconv = (NHID * KPAD + THREADS - 1) / THREADS;
        conv_w1_bf16<<<dim3(nconv), block, 0, stream>>>(W1, W1b);
        sparse_mnist_fused<true><<<grid, block, 0, stream>>>(
            x, W1, W1b, b1, W2, b2, duty, out);
    } else {
        sparse_mnist_fused<false><<<grid, block, 0, stream>>>(
            x, W1, nullptr, b1, W2, b2, duty, out);
    }
}
